// OuterProductNetwork_11974368821314
// MI455X (gfx1250) — matrix operand order repaired and checked by reading, not compiled, not run on hardware
//
#include <hip/hip_runtime.h>

#define FIELDS 32
#define EMBED  64
#define PAIRS  496   // 32*31/2
#define BATCH  2048

#define BT   128     // batch rows per block
#define KS   72      // LDS row stride (bf16 elems) for K_p, padded vs bank conflicts
#define QS   72      // LDS row stride (bf16 elems) for Q
#define PS   68      // LDS row stride (f32 elems) for P

typedef __attribute__((ext_vector_type(16))) __bf16 v16bf;
typedef __attribute__((ext_vector_type(8)))  float  v8f;

union FragBF {
    unsigned int u[8];
    v16bf        v;
};

__device__ __forceinline__ unsigned short f2bf(float f) {
    // round-to-nearest-even fp32 -> bf16 (NaN edge ignored; inputs are finite)
    unsigned int u = __float_as_uint(f);
    u += 0x7fffu + ((u >> 16) & 1u);
    return (unsigned short)(u >> 16);
}

__global__ __launch_bounds__(256)
void opn_wmma_kernel(const float* __restrict__ x,
                     const float* __restrict__ kernel,
                     float* __restrict__ out) {
    // blockIdx.x : batch tile (BATCH/BT = 16), blockIdx.y : pair p (496)
    const int p  = blockIdx.y;
    const int b0 = blockIdx.x * BT;

    // pair p -> (fi, fj) of triu_indices(32, k=1): row-major upper triangle
    int pp = p, fi = 0;
    while (pp >= (FIELDS - 1) - fi) { pp -= (FIELDS - 1) - fi; ++fi; }
    const int fj = fi + 1 + pp;

    __shared__ unsigned short Ks[EMBED * KS];  // K_p transposed: Ks[e][f] = kernel[f,p,e] (bf16)
    __shared__ unsigned short Qs[BT * QS];     // Qs[b_local][f] = x[b, fj, f]          (bf16)
    __shared__ float          Psh[BT * PS];    // Psh[b_local][e] = x[b, fi, e]          (f32)

    const int tid = threadIdx.x;

    // ---- stage K_p into LDS (coalesced read, transposed bf16 store) ----
    #pragma unroll 4
    for (int idx = tid; idx < EMBED * EMBED; idx += 256) {
        const int f = idx >> 6;        // 0..63
        const int e = idx & 63;        // 0..63
        const float v = kernel[((size_t)f * PAIRS + p) * EMBED + e];
        Ks[e * KS + f] = f2bf(v);
    }

    // ---- stage P (f32) and Q (bf16) rows for this batch tile ----
    #pragma unroll 4
    for (int idx = tid; idx < BT * EMBED; idx += 256) {
        const int bl = idx >> 6;       // 0..127
        const int f  = idx & 63;       // 0..63
        const int b  = b0 + bl;
        Qs[bl * QS + f]  = f2bf(x[((size_t)b * FIELDS + fj) * EMBED + f]);
        Psh[bl * PS + f] =       x[((size_t)b * FIELDS + fi) * EMBED + f];
    }

    __syncthreads();

    // ---- per-wave 16-batch sub-tile: T = A(e x f) * B(f x b), bf16 WMMA, f32 acc ----
    const int lane = tid & 31;
    const int wv   = tid >> 5;          // 0..7
    const int hi   = lane >> 4;         // lane half
    const int m    = lane & 15;         // A row (e_local) / B,D column (b_local offset)
    const int bl   = wv * 16 + m;       // this lane's batch row for B/D

    float s = 0.0f;

    #pragma unroll
    for (int eb = 0; eb < 4; ++eb) {    // e blocks of 16
        v8f acc = {};
        #pragma unroll
        for (int fb = 0; fb < EMBED; fb += 32) {   // K blocks of 32
            FragBF a, b;
            #pragma unroll
            for (int v = 0; v < 8; ++v) {
                // 16-bit A 16x32 layout: VGPR v holds K pair; lane halves interleave
                // lanes 0-15: K in {0-7,16-23}; lanes 16-31: K in {8-15,24-31}
                const int fA = (v < 4) ? (hi * 8 + 2 * v)
                                       : (16 + hi * 8 + 2 * (v - 4));
                a.u[v] = *(const unsigned int*)&Ks[(eb * 16 + m) * KS + fb + fA];
                // 16-bit B 32x16 layout: same K interleave as A
                const int fB = fA;
                b.u[v] = *(const unsigned int*)&Qs[bl * QS + fb + fB];
            }
            acc = __builtin_amdgcn_wmma_f32_16x16x32_bf16(
                      /*neg_a=*/false, a.v, /*neg_b=*/false, b.v,
                      /*c_mod=*/(short)0, acc,
                      /*reuse_a=*/false, /*reuse_b=*/false);
        }
        // D[e_row, b_col]: lane holds b = lane%16, e = eb*16 + hi*8 + j (VGPR j)
        #pragma unroll
        for (int j = 0; j < 8; ++j) {
            const int e = eb * 16 + hi * 8 + j;
            s += acc[j] * Psh[bl * PS + e];
        }
    }

    // combine the two lane halves (e subsets) for the same batch row b
    s += __shfl_xor(s, 16, 32);

    if (hi == 0) {
        out[(size_t)(b0 + bl) * PAIRS + p] = s;
    }
}

extern "C" void kernel_launch(void* const* d_in, const int* in_sizes, int n_in,
                              void* d_out, int out_size, void* d_ws, size_t ws_size,
                              hipStream_t stream) {
    (void)in_sizes; (void)n_in; (void)out_size; (void)d_ws; (void)ws_size;
    const float* x      = (const float*)d_in[0];   // (2048, 32, 64) f32
    const float* kernel = (const float*)d_in[1];   // (64, 496, 64)  f32
    float* out          = (float*)d_out;           // (2048, 496)    f32

    dim3 grid(BATCH / BT, PAIRS);   // (16, 496)
    dim3 block(256);                // 8 waves (wave32)
    opn_wmma_kernel<<<grid, block, 0, stream>>>(x, kernel, out);
}
